// HypergraphNet_44349832298680
// MI455X (gfx1250) — matrix operand order repaired and checked by reading, not compiled, not run on hardware
//
#include <hip/hip_runtime.h>

#define NN   50000
#define NE   4000
#define NNZV 400000
#define NG   64
#define CH   1024

typedef __attribute__((ext_vector_type(16))) __bf16 v16bf;
typedef __attribute__((ext_vector_type(8)))  __bf16 v8bf;
typedef __attribute__((ext_vector_type(8)))  float  v8f;

__device__ inline unsigned short f2bf_us(float f) {
    union { float f; unsigned int u; } a; a.f = f;
    unsigned int u = a.u;
    unsigned int r = u + 0x7FFFu + ((u >> 16) & 1u);   // round-to-nearest-even
    return (unsigned short)(r >> 16);
}

// ---------------- utility kernels ----------------

__global__ void k_zero_i(int* __restrict__ p, int n) {
    int i = blockIdx.x * blockDim.x + threadIdx.x;
    if (i < n) p[i] = 0;
}

__global__ void k_count(const int* __restrict__ ni, const int* __restrict__ hi,
                        int* __restrict__ dcnt, int* __restrict__ bcnt, int nnz) {
    int i = blockIdx.x * blockDim.x + threadIdx.x;
    if (i < nnz) {
        atomicAdd(&dcnt[ni[i]], 1);
        atomicAdd(&bcnt[hi[i]], 1);
    }
}

// exclusive scan, single block of 1024 threads, arbitrary length
__global__ __launch_bounds__(1024) void k_scan_excl(const int* __restrict__ in,
                                                    int* __restrict__ out, int L) {
    __shared__ int sd[1024];
    int carry = 0;
    for (int base = 0; base < L; base += 1024) {
        int i = base + (int)threadIdx.x;
        int v = (i < L) ? in[i] : 0;
        sd[threadIdx.x] = v;
        __syncthreads();
        for (int off = 1; off < 1024; off <<= 1) {
            int t = (threadIdx.x >= (unsigned)off) ? sd[threadIdx.x - off] : 0;
            __syncthreads();
            sd[threadIdx.x] += t;
            __syncthreads();
        }
        int incl = sd[threadIdx.x];
        if (i < L) out[i] = carry + incl - v;
        int total = sd[1023];
        __syncthreads();
        carry += total;
    }
}

__global__ void k_fill(const int* __restrict__ ni, const int* __restrict__ hi,
                       const int* __restrict__ eoff, const int* __restrict__ noff,
                       int* __restrict__ ecur, int* __restrict__ ncur,
                       int* __restrict__ edge_nodes, int* __restrict__ node_edges, int nnz) {
    int i = blockIdx.x * blockDim.x + threadIdx.x;
    if (i >= nnz) return;
    int n = ni[i], e = hi[i];
    int p = atomicAdd(&ecur[e], 1);
    edge_nodes[eoff[e] + p] = n;
    int q = atomicAdd(&ncur[n], 1);
    node_edges[noff[n] + q] = e;
}

__global__ void k_inv(const int* __restrict__ cnt, float* __restrict__ inv, int n) {
    int i = blockIdx.x * blockDim.x + threadIdx.x;
    if (i < n) inv[i] = cnt[i] > 0 ? 1.0f / (float)cnt[i] : 0.0f;
}

// transpose + bf16 cast: Wt[n*CH + k] = bf16(W[k*CH + n])
__global__ void k_wt(const float* __restrict__ W, unsigned short* __restrict__ Wt) {
    int idx = blockIdx.x * blockDim.x + threadIdx.x;
    if (idx >= CH * CH) return;
    int n = idx >> 10;
    int k = idx & (CH - 1);
    Wt[idx] = f2bf_us(W[k * CH + n]);
}

__global__ void k_cast(const float* __restrict__ in, unsigned short* __restrict__ out, int n) {
    int i = blockIdx.x * blockDim.x + threadIdx.x;
    if (i < n) out[i] = f2bf_us(in[i]);
}

// ---------------- bf16 WMMA GEMM: C[M,Nc] = A[M,K] * Bt[Nc,K]^T ----------------
// block: 256 threads = 8 waves; wave tile 32x32; block tile 128x64

union Frag { v16bf v; v8bf h[2]; };

__device__ inline void store_tile(v8f acc, float* __restrict__ C, int M, int Nc,
                                  int mBase, int nBase, int half, int lr) {
    int col = nBase + lr;
#pragma unroll
    for (int j = 0; j < 8; ++j) {
        int row = mBase + j + half * 8;
        if (row < M) C[(size_t)row * Nc + col] = acc[j];
    }
}

__global__ __launch_bounds__(256) void k_gemm(const unsigned short* __restrict__ A,
                                              const unsigned short* __restrict__ Bt,
                                              float* __restrict__ C,
                                              int M, int K, int Nc) {
    const int lane = threadIdx.x & 31;
    const int wave = threadIdx.x >> 5;
    const int wm = wave & 3;
    const int wn = wave >> 2;
    const int m0 = blockIdx.x * 128 + wm * 32;
    const int n0 = blockIdx.y * 64 + wn * 32;
    const int half = lane >> 4;
    const int lr = lane & 15;

    // A rows (clamped so all lanes stay in bounds; stores are guarded)
    int ra0 = m0 + lr;       if (ra0 >= M) ra0 = M - 1;
    int ra1 = m0 + 16 + lr;  if (ra1 >= M) ra1 = M - 1;
    const unsigned short* pa0 = A + (size_t)ra0 * K + half * 8;   // K chunks {h*8, 16+h*8}
    const unsigned short* pa1 = A + (size_t)ra1 * K + half * 8;
    const unsigned short* pb0 = Bt + (size_t)(n0 + lr) * K + half * 8;       // K chunks {h*8, 16+h*8}
    const unsigned short* pb1 = Bt + (size_t)(n0 + 16 + lr) * K + half * 8;

    v8f acc00 = {}, acc01 = {}, acc10 = {}, acc11 = {};

    for (int k = 0; k < K; k += 32) {
        __builtin_prefetch(pa0 + k + 64, 0, 0);
        __builtin_prefetch(pa1 + k + 64, 0, 0);
        Frag a0, a1, b0, b1;
        a0.h[0] = *(const v8bf*)(pa0 + k);
        a0.h[1] = *(const v8bf*)(pa0 + k + 16);
        a1.h[0] = *(const v8bf*)(pa1 + k);
        a1.h[1] = *(const v8bf*)(pa1 + k + 16);
        b0.h[0] = *(const v8bf*)(pb0 + k);
        b0.h[1] = *(const v8bf*)(pb0 + k + 16);
        b1.h[0] = *(const v8bf*)(pb1 + k);
        b1.h[1] = *(const v8bf*)(pb1 + k + 16);
        acc00 = __builtin_amdgcn_wmma_f32_16x16x32_bf16(false, a0.v, false, b0.v, (short)0, acc00, false, false);
        acc01 = __builtin_amdgcn_wmma_f32_16x16x32_bf16(false, a0.v, false, b1.v, (short)0, acc01, false, false);
        acc10 = __builtin_amdgcn_wmma_f32_16x16x32_bf16(false, a1.v, false, b0.v, (short)0, acc10, false, false);
        acc11 = __builtin_amdgcn_wmma_f32_16x16x32_bf16(false, a1.v, false, b1.v, (short)0, acc11, false, false);
    }
    store_tile(acc00, C, M, Nc, m0,      n0,      half, lr);
    store_tile(acc01, C, M, Nc, m0,      n0 + 16, half, lr);
    store_tile(acc10, C, M, Nc, m0 + 16, n0,      half, lr);
    store_tile(acc11, C, M, Nc, m0 + 16, n0 + 16, half, lr);
}

// ---------------- sparse aggregation (CSR gathers) ----------------

// m[e,c] = Binv[e] * sum_{n in edge} hW[n,c]
__global__ void k_edge_gather(const float* __restrict__ hW, const int* __restrict__ edge_nodes,
                              const int* __restrict__ eoff, const int* __restrict__ bcnt,
                              const float* __restrict__ Binv, float* __restrict__ m) {
    int e = blockIdx.x;
    int c = blockIdx.y * blockDim.x + threadIdx.x;
    int s = eoff[e], t = s + bcnt[e];
    float sum = 0.0f;
    for (int j = s; j < t; ++j) {
        int n = edge_nodes[j];
        sum += hW[(size_t)n * CH + c];
    }
    m[(size_t)e * CH + c] = sum * Binv[e];
}

// out[n,c] = Dinv[n] * sum_{e in node} m[e,c] + bias[c]
__global__ void k_node_gather(const float* __restrict__ m, const int* __restrict__ node_edges,
                              const int* __restrict__ noff, const int* __restrict__ dcnt,
                              const float* __restrict__ Dinv, const float* __restrict__ bias,
                              float* __restrict__ out) {
    int n = blockIdx.x;
    int c = blockIdx.y * blockDim.x + threadIdx.x;
    int s = noff[n], t = s + dcnt[n];
    float sum = 0.0f;
    for (int j = s; j < t; ++j) {
        int e = node_edges[j];
        sum += m[(size_t)e * CH + c];
    }
    out[(size_t)n * CH + c] = sum * Dinv[n] + bias[c];
}

// ---------------- batch norm ----------------

__global__ void k_bn_stats(const float* __restrict__ h, float* __restrict__ stats, int Nn) {
    int c = blockIdx.x * blockDim.x + threadIdx.x;   // channel, grid exact
    int r0 = blockIdx.y * 512;
    int rend = r0 + 512; if (rend > Nn) rend = Nn;
    float s = 0.0f, q = 0.0f;
    for (int r = r0; r < rend; ++r) {
        float v = h[(size_t)r * CH + c];
        s += v;
        q += v * v;
    }
    atomicAdd(&stats[c], s);
    atomicAdd(&stats[CH + c], q);
}

__global__ void k_bn_apply(float* __restrict__ h, unsigned short* __restrict__ hbf,
                           const float* __restrict__ stats, const float* __restrict__ gamma,
                           const float* __restrict__ beta, int Nn) {
    size_t i = (size_t)blockIdx.x * blockDim.x + threadIdx.x;
    size_t total = (size_t)Nn * CH;
    if (i >= total) return;
    int c = (int)(i & (CH - 1));
    float invN = 1.0f / (float)Nn;
    float mu = stats[c] * invN;
    float var = stats[CH + c] * invN - mu * mu;
    float x = h[i];
    float y = (x - mu) * rsqrtf(var + 1e-5f) * gamma[c] + beta[c];
    y = fmaxf(y, 0.0f);
    h[i] = y;
    hbf[i] = f2bf_us(y);
}

// ---------------- pooling + FC ----------------

__global__ void k_pool(const float* __restrict__ h, const int* __restrict__ batch,
                       float* __restrict__ pooled, int* __restrict__ gcnt) {
    int n = blockIdx.x;
    int c = blockIdx.y * blockDim.x + threadIdx.x;
    int g = batch[n];
    atomicAdd(&pooled[(size_t)g * CH + c], h[(size_t)n * CH + c]);
    if (c == 0) atomicAdd(&gcnt[g], 1);
}

__global__ __launch_bounds__(256) void k_fc(const float* __restrict__ pooled,
                                            const int* __restrict__ gcnt,
                                            const float* __restrict__ Wfc,
                                            const float* __restrict__ bfc,
                                            float* __restrict__ out) {
    __shared__ float red[256];
    int g = blockIdx.x;
    float s = 0.0f;
    for (int c = threadIdx.x; c < CH; c += 256)
        s += pooled[(size_t)g * CH + c] * Wfc[c];
    red[threadIdx.x] = s;
    __syncthreads();
    for (int off = 128; off > 0; off >>= 1) {
        if ((int)threadIdx.x < off) red[threadIdx.x] += red[threadIdx.x + off];
        __syncthreads();
    }
    if (threadIdx.x == 0) {
        float invc = 1.0f / fmaxf((float)gcnt[g], 1.0f);
        float v = red[0] * invc + bfc[0];
        out[g] = 1.0f / (1.0f + expf(-v));
    }
}

// ---------------- launcher ----------------

extern "C" void kernel_launch(void* const* d_in, const int* in_sizes, int n_in,
                              void* d_out, int out_size, void* d_ws, size_t ws_size,
                              hipStream_t stream) {
    const float* x             = (const float*)d_in[0];
    const int*   node_idx      = (const int*)d_in[1];
    const int*   hyperedge_idx = (const int*)d_in[2];
    const int*   batch         = (const int*)d_in[3];
    const float* W[3]  = { (const float*)d_in[4], (const float*)d_in[6], (const float*)d_in[8] };
    const float* b[3]  = { (const float*)d_in[5], (const float*)d_in[7], (const float*)d_in[9] };
    const float* g[3]  = { (const float*)d_in[10], (const float*)d_in[12], (const float*)d_in[14] };
    const float* be[3] = { (const float*)d_in[11], (const float*)d_in[13], (const float*)d_in[15] };
    const float* Wfc = (const float*)d_in[16];
    const float* bfc = (const float*)d_in[17];
    float* out = (float*)d_out;

    char* p = (char*)d_ws;
    auto alloc = [&](size_t bytes) -> char* {
        char* r = p;
        p += (bytes + 255) & ~(size_t)255;
        return r;
    };
    float*          Q          = (float*)         alloc((size_t)NN * CH * 4);
    unsigned short* hbf        = (unsigned short*)alloc((size_t)NN * CH * 2);
    float*          m          = (float*)         alloc((size_t)NE * CH * 4);
    unsigned short* Wt         = (unsigned short*)alloc((size_t)3 * CH * CH * 2);
    int*            edge_nodes = (int*)           alloc((size_t)NNZV * 4);
    int*            node_edges = (int*)           alloc((size_t)NNZV * 4);
    int*            eoff       = (int*)           alloc((size_t)NE * 4);
    int*            noff       = (int*)           alloc((size_t)NN * 4);
    int*            dcnt       = (int*)           alloc((size_t)NN * 4);
    int*            bcnt       = (int*)           alloc((size_t)NE * 4);
    int*            ecur       = (int*)           alloc((size_t)NE * 4);
    int*            ncur       = (int*)           alloc((size_t)NN * 4);
    float*          Dinv       = (float*)         alloc((size_t)NN * 4);
    float*          Binv       = (float*)         alloc((size_t)NE * 4);
    float*          stats      = (float*)         alloc((size_t)2 * CH * 4);
    float*          pooled     = (float*)         alloc((size_t)NG * CH * 4);
    int*            gcnt       = (int*)           alloc((size_t)NG * 4);

    const int B256 = 256;
    // --- CSR build (indices are launch-invariant but rebuilt each call: no cross-call state) ---
    k_zero_i<<<(NN + 255) / 256, B256, 0, stream>>>(dcnt, NN);
    k_zero_i<<<(NE + 255) / 256, B256, 0, stream>>>(bcnt, NE);
    k_zero_i<<<(NN + 255) / 256, B256, 0, stream>>>(ncur, NN);
    k_zero_i<<<(NE + 255) / 256, B256, 0, stream>>>(ecur, NE);
    k_count<<<(NNZV + 255) / 256, B256, 0, stream>>>(node_idx, hyperedge_idx, dcnt, bcnt, NNZV);
    k_scan_excl<<<1, 1024, 0, stream>>>(bcnt, eoff, NE);
    k_scan_excl<<<1, 1024, 0, stream>>>(dcnt, noff, NN);
    k_fill<<<(NNZV + 255) / 256, B256, 0, stream>>>(node_idx, hyperedge_idx, eoff, noff,
                                                    ecur, ncur, edge_nodes, node_edges, NNZV);
    k_inv<<<(NN + 255) / 256, B256, 0, stream>>>(dcnt, Dinv, NN);
    k_inv<<<(NE + 255) / 256, B256, 0, stream>>>(bcnt, Binv, NE);

    // --- weights to transposed bf16 ---
    for (int w = 0; w < 3; ++w)
        k_wt<<<(CH * CH + 255) / 256, B256, 0, stream>>>(W[w], Wt + (size_t)w * CH * CH);

    // --- layer input cast ---
    k_cast<<<(NN * CH + 255) / 256, B256, 0, stream>>>(x, hbf, NN * CH);

    dim3 gemmGrid((NN + 127) / 128, CH / 64);
    dim3 edgeGrid(NE, CH / 256);
    dim3 nodeGrid(NN, CH / 256);
    dim3 statGrid(CH / 256, (NN + 511) / 512);
    int applyBlocks = (int)(((size_t)NN * CH + 255) / 256);

    for (int l = 0; l < 3; ++l) {
        k_gemm<<<gemmGrid, B256, 0, stream>>>(hbf, Wt + (size_t)l * CH * CH, Q, NN, CH, CH);
        k_edge_gather<<<edgeGrid, B256, 0, stream>>>(Q, edge_nodes, eoff, bcnt, Binv, m);
        k_node_gather<<<nodeGrid, B256, 0, stream>>>(m, node_edges, noff, dcnt, Dinv, b[l], Q);
        k_zero_i<<<(2 * CH + 255) / 256, B256, 0, stream>>>((int*)stats, 2 * CH);
        k_bn_stats<<<statGrid, B256, 0, stream>>>(Q, stats, NN);
        k_bn_apply<<<applyBlocks, B256, 0, stream>>>(Q, hbf, stats, g[l], be[l], NN);
    }

    // --- pooling + FC + sigmoid ---
    k_zero_i<<<(NG * CH + 255) / 256, B256, 0, stream>>>((int*)pooled, NG * CH);
    k_zero_i<<<1, B256, 0, stream>>>(gcnt, NG);
    k_pool<<<nodeGrid, B256, 0, stream>>>(Q, batch, pooled, gcnt);
    k_fc<<<NG, B256, 0, stream>>>(pooled, gcnt, Wfc, bfc, out);
}
